// StructureAwareCFGenerator_11175504904347
// MI455X (gfx1250) — matrix operand order repaired and checked by reading, not compiled, not run on hardware
//
#include <hip/hip_runtime.h>
#include <hip/hip_bf16.h>

typedef __attribute__((ext_vector_type(16))) _Float16 v16h;
typedef __attribute__((ext_vector_type(8)))  float    v8f;
typedef __attribute__((ext_vector_type(2)))  float    v2f;
typedef __attribute__((ext_vector_type(2)))  _Float16 v2h;

// Problem constants (from reference setup_inputs)
#define BB 4096
#define TT 512
#define SS 12
#define AA 2
#define VV 5
#define H1 64
#define H2 32
#define XK 17               // S + V (real K of layer 1, padded to 32)
#define INV_VALUE 1.0f
#define CORR_SCALE 0.1f
#define EDGE_THRESH 0.1f
#define MAX_CHANGE 0.5f

#define WAVES_PER_BLOCK 8   // 256 threads, wave32
#define TILE_TOK 16
#define TPW 16              // tiles per wave (amortize weight fragments)

// A-fragment K order used by this code (contiguous, not the ISA table order; 16-bit A 16x32):
//   lane = row + 16*half2 ; element e holds K = e + 16*half2
// Inverse (given k in [0,32)): half2 = k>>4 ; e = k&15
__device__ __forceinline__ int frag_half(int k) { return k >> 4; }
__device__ __forceinline__ int frag_elem(int k) { return k & 15; }

// Pack two f32 to f16 (v_cvt_pk_f16_f32) then ReLU in f16 (v_pk_max_num_f16).
__device__ __forceinline__ v2h relu_pk(float a, float b)
{
    v2f pf = {a, b};
    v2h ph = __builtin_convertvector(pf, v2h);
    v2h z  = {};
    return __builtin_elementwise_max(ph, z);
}

// ---------------------------------------------------------------------------
// Kernel 1: intervention + causal blend + 3-layer MLP via WMMA.
// ---------------------------------------------------------------------------
__global__ __launch_bounds__(256) void cf_mlp_wmma(
    const float* __restrict__ base_states,   // [B*T, S]
    const float* __restrict__ base_causal,   // [B*T, V]
    const float* __restrict__ graph,         // [V, V]
    const float* __restrict__ W1, const float* __restrict__ b1,
    const float* __restrict__ W2, const float* __restrict__ b2,
    const float* __restrict__ W3, const float* __restrict__ b3,
    const int*   __restrict__ target_ptr,
    float* __restrict__ out_states,          // [B*T, S] raw (scan applied later)
    float* __restrict__ out_cf)              // [B*T, V]
{
    // Block-shared weight fragments (identical for all waves); built once.
    __shared__ __align__(16) _Float16 wfrag[9][32][16];                 // 9216 B
    // Per-wave scratch, all in WMMA fragment layout where applicable.
    __shared__ __align__(16) float    xsS  [WAVES_PER_BLOCK][16][12];   // 6144 B
    __shared__ __align__(16) _Float16 afrag[WAVES_PER_BLOCK][32][16];   // 8192 B
    __shared__ __align__(16) _Float16 h1f  [WAVES_PER_BLOCK][2][32][16];// 16384 B
    __shared__ __align__(16) _Float16 h2f  [WAVES_PER_BLOCK][32][16];   // 8192 B

    const int tid  = threadIdx.x;
    const int w    = tid >> 5;
    const int lane = tid & 31;
    const int half = lane >> 4;
    const int m    = lane & 15;

    // ---- Cooperative weight-fragment build (branch-free masking) ----------
    // Layer 1: frags 0..3  (B layout: elem e -> K = e + 16*half)
    for (int i = tid; i < 4 * 512; i += 256) {
        const int f = i >> 9, ln = (i >> 4) & 31, e = i & 15;
        const int k = e + 16 * (ln >> 4);
        const int col = f * 16 + (ln & 15);
        const float v = W1[(k < XK ? k : XK - 1) * H1 + col] * (k < XK ? 1.0f : 0.0f);
        wfrag[f][ln][e] = (_Float16)v;
    }
    // Layer 2: frags 4..7  (4,5: K-block 0, cols 0/16 ; 6,7: K-block 1)
    for (int i = tid; i < 4 * 512; i += 256) {
        const int f = i >> 9, ln = (i >> 4) & 31, e = i & 15;
        const int kblk = f >> 1, j = f & 1;
        const int k = kblk * 32 + e + 16 * (ln >> 4);
        wfrag[4 + f][ln][e] = (_Float16)W2[k * H2 + j * 16 + (ln & 15)];
    }
    // Layer 3: frag 8 (cols padded 12 -> 16)
    for (int i = tid; i < 512; i += 256) {
        const int ln = (i >> 4) & 31, e = i & 15;
        const int k = e + 16 * (ln >> 4);
        const int mm = ln & 15;
        const float v = W3[k * SS + (mm < SS ? mm : SS - 1)] * (mm < SS ? 1.0f : 0.0f);
        wfrag[8][ln][e] = (_Float16)v;
    }
    __syncthreads();

    // ---- Per-wave register copies of the fragments (2x ds_load_b128 each) -
    v16h B1f[4], B2lo[2], B2hi[2], B3f;
#pragma unroll
    for (int j = 0; j < 4; ++j) B1f[j] = *(const v16h*)&wfrag[j][lane][0];
#pragma unroll
    for (int j = 0; j < 2; ++j) {
        B2lo[j] = *(const v16h*)&wfrag[4 + j][lane][0];
        B2hi[j] = *(const v16h*)&wfrag[6 + j][lane][0];
    }
    B3f = *(const v16h*)&wfrag[8][lane][0];

    // ---- Per-lane constants hoisted out of the tile loop ------------------
    const int tv = *target_ptr;
    float bias1[4], bias2[2];
#pragma unroll
    for (int j = 0; j < 4; ++j) bias1[j] = b1[j * 16 + m];
#pragma unroll
    for (int j = 0; j < 2; ++j) bias2[j] = b2[j * 16 + m];
    const float bias3 = (m < SS) ? b3[m] : 0.0f;

    // Zero afrag once; K = 17..31 positions stay zero across all tiles.
    {
        v16h z = {};
        *(v16h*)&afrag[w][lane][0] = z;
    }

    const long tileBase = ((long)blockIdx.x * WAVES_PER_BLOCK + w) * TPW;

    for (int it = 0; it < TPW; ++it) {
        const long tokBase = (tileBase + it) * TILE_TOK;

        // ---- Stage states: 16 tokens x 12 floats = 48 float4 chunks -------
        for (int i = lane; i < 48; i += 32) {
            const int tl = i / 3, c = i % 3;             // k = 4c .. 4c+3
            const float4 sv = *(const float4*)(base_states + (tokBase + tl) * SS + c * 4);
            *(float4*)&xsS[w][tl][c * 4] = sv;
            const int k0 = c * 4;
            const int h2 = frag_half(k0);                // whole chunk same half
            const int e0 = frag_elem(k0);                // e contiguous for chunk
            _Float16* dst = &afrag[w][tl + 16 * h2][e0];
            dst[0] = (_Float16)sv.x; dst[1] = (_Float16)sv.y;
            dst[2] = (_Float16)sv.z; dst[3] = (_Float16)sv.w;
        }
        // ---- Stage causal: intervention + graph blend, emit cf ------------
        for (int i = lane; i < 16 * VV; i += 32) {
            const int tl = i / VV, v = i % VV;
            const long tok = tokBase + tl;
            const int  t   = (int)(tok & (TT - 1));
            const float raw = base_causal[tok * VV + v];
            const float g   = graph[tv * VV + v];
            float cfv;
            if (v == tv)                        cfv = INV_VALUE;
            else if (g > EDGE_THRESH && t >= 1) cfv = 0.7f * raw + 0.3f * g * INV_VALUE;
            else                                cfv = raw;
            const int k = SS + v;                        // 12..16
            afrag[w][tl + 16 * frag_half(k)][frag_elem(k)] = (_Float16)cfv;
            out_cf[tok * VV + v] = cfv;
        }
        asm volatile("s_wait_dscnt 0" ::: "memory");

        // ---- Layer 1: X(16x32) x W1 -> H1(16x64) --------------------------
        const v16h a1 = *(const v16h*)&afrag[w][lane][0];
#pragma unroll
        for (int j = 0; j < 4; ++j) {
            const float bs = bias1[j];
            v8f c = {bs, bs, bs, bs, bs, bs, bs, bs};
            v8f acc = __builtin_amdgcn_wmma_f32_16x16x32_f16(
                false, a1, false, B1f[j], (short)0, c, false, false);
            const int col = j * 16 + m;                  // hidden index 0..63
            const int fh = col >> 5, kk = col & 31;
            const int h2 = frag_half(kk), e2 = frag_elem(kk);
#pragma unroll
            for (int r = 0; r < 8; r += 2) {
                const int row = r + 8 * half;
                const v2h ph = relu_pk(acc[r], acc[r + 1]);
                h1f[w][fh][row     + 16 * h2][e2] = ph[0];
                h1f[w][fh][row + 1 + 16 * h2][e2] = ph[1];
            }
        }
        asm volatile("s_wait_dscnt 0" ::: "memory");

        // ---- Layer 2: H1(16x64) x W2 -> H2(16x32), K chained 32+32 --------
        const v16h a2lo = *(const v16h*)&h1f[w][0][lane][0];
        const v16h a2hi = *(const v16h*)&h1f[w][1][lane][0];
#pragma unroll
        for (int j = 0; j < 2; ++j) {
            const float bs = bias2[j];
            v8f c = {bs, bs, bs, bs, bs, bs, bs, bs};
            v8f acc = __builtin_amdgcn_wmma_f32_16x16x32_f16(
                false, a2lo, false, B2lo[j], (short)0, c, false, false);
            acc = __builtin_amdgcn_wmma_f32_16x16x32_f16(
                false, a2hi, false, B2hi[j], (short)0, acc, false, false);
            const int kk = j * 16 + m;                   // hidden index 0..31
            const int h2 = frag_half(kk), e2 = frag_elem(kk);
#pragma unroll
            for (int r = 0; r < 8; r += 2) {
                const int row = r + 8 * half;
                const v2h ph = relu_pk(acc[r], acc[r + 1]);
                h2f[w][row     + 16 * h2][e2] = ph[0];
                h2f[w][row + 1 + 16 * h2][e2] = ph[1];
            }
        }
        asm volatile("s_wait_dscnt 0" ::: "memory");

        // ---- Layer 3: H2(16x32) x W3(32x16 padded) ------------------------
        const v16h a3 = *(const v16h*)&h2f[w][lane][0];
        v8f c3 = {bias3, bias3, bias3, bias3, bias3, bias3, bias3, bias3};
        const v8f acc3 = __builtin_amdgcn_wmma_f32_16x16x32_f16(
            false, a3, false, B3f, (short)0, c3, false, false);

        // ---- Epilogue: cf_states_raw = base_states + 0.1 * mlp ------------
        if (m < SS) {
#pragma unroll
            for (int r = 0; r < 8; ++r) {
                const int  row = r + 8 * half;
                const long tok = tokBase + row;
                out_states[tok * SS + m] = xsS[w][row][m] + CORR_SCALE * acc3[r];
            }
        }
    }
}

// ---------------------------------------------------------------------------
// Kernel 2: in-place sequential consistency scan over T per (b, s) chain.
// ---------------------------------------------------------------------------
__global__ __launch_bounds__(256) void consistency_scan(float* __restrict__ st)
{
    const int tid = blockIdx.x * 256 + threadIdx.x;   // 0 .. B*S-1
    const int b = tid / SS;
    const int s = tid % SS;
    float prev = st[((long)b * TT) * SS + s];         // t = 0 unchanged
    for (int t = 1; t < TT; ++t) {
        const long idx = ((long)b * TT + t) * SS + s;
        const float cur = st[idx];
        float d = cur - prev;
        d = fminf(MAX_CHANGE, fmaxf(-MAX_CHANGE, d));
        prev += d;
        st[idx] = prev;
    }
}

// ---------------------------------------------------------------------------
// Kernel 3: cf_actions = base_actions (identity), vectorized copy.
// ---------------------------------------------------------------------------
__global__ __launch_bounds__(256) void copy_actions(
    const float4* __restrict__ src, float4* __restrict__ dst, int n4)
{
    const int i = blockIdx.x * 256 + threadIdx.x;
    if (i < n4) dst[i] = src[i];
}

extern "C" void kernel_launch(void* const* d_in, const int* in_sizes, int n_in,
                              void* d_out, int out_size, void* d_ws, size_t ws_size,
                              hipStream_t stream)
{
    (void)in_sizes; (void)n_in; (void)d_ws; (void)ws_size; (void)out_size;

    const float* base_states  = (const float*)d_in[0];
    const float* base_actions = (const float*)d_in[1];
    const float* base_causal  = (const float*)d_in[2];
    const float* causal_graph = (const float*)d_in[3];
    const float* W1 = (const float*)d_in[4];
    const float* b1 = (const float*)d_in[5];
    const float* W2 = (const float*)d_in[6];
    const float* b2 = (const float*)d_in[7];
    const float* W3 = (const float*)d_in[8];
    const float* b3 = (const float*)d_in[9];
    const int*   tv = (const int*)d_in[10];

    float* out = (float*)d_out;
    float* out_states  = out;                                   // B*T*S
    float* out_actions = out + (long)BB * TT * SS;              // B*T*A
    float* out_cf      = out_actions + (long)BB * TT * AA;      // B*T*V

    // 131072 tiles / (8 waves * 16 tiles) = 1024 blocks.
    const int nTiles  = (BB * TT) / TILE_TOK;
    const int nBlocks = nTiles / (WAVES_PER_BLOCK * TPW);
    cf_mlp_wmma<<<nBlocks, 256, 0, stream>>>(
        base_states, base_causal, causal_graph,
        W1, b1, W2, b2, W3, b3, tv, out_states, out_cf);

    consistency_scan<<<(BB * SS) / 256, 256, 0, stream>>>(out_states);

    const int n4 = (BB * TT * AA) / 4;
    copy_actions<<<(n4 + 255) / 256, 256, 0, stream>>>(
        (const float4*)base_actions, (float4*)out_actions, n4);
}
